// QuantumAttentionGate_16106127360117
// MI455X (gfx1250) — compile-verified, matrix operand order checked by reading, not run on hardware
//
#include <hip/hip_runtime.h>
#include <math.h>

typedef __attribute__((ext_vector_type(16))) _Float16 v16h;
typedef __attribute__((ext_vector_type(8)))  float    v8f;

#define NQ      6
#define NL      2
#define BATCH   32768
#define IN_DIM  4096
#define NH      32
#define QDIM    64

#define WAVES_PER_BLOCK   2
#define THREADS_PER_BLOCK (WAVES_PER_BLOCK * 32)
#define ROWS_PER_WAVE     32
#define ROWS_PER_BLOCK    (WAVES_PER_BLOCK * ROWS_PER_WAVE)

// branch-free tanh: 1 - 2/(exp(2x)+1); exact saturation at +/-1 via inf semantics
__device__ __forceinline__ float fast_tanhf(float x) {
    const float e = __expf(2.0f * x);
    return 1.0f - 2.0f / (e + 1.0f);
}

__global__ __launch_bounds__(THREADS_PER_BLOCK) void
qagate_fused_kernel(const float* __restrict__ sv,     // (B, IN_DIM)
                    const float* __restrict__ Wang,   // (NQ, IN_DIM)
                    const float* __restrict__ bang,   // (NQ,)
                    const float* __restrict__ Wpost,  // (NH, NQ)
                    const float* __restrict__ bpost,  // (NH,)
                    const float* __restrict__ qw,     // (NL, NQ, 3)
                    float* __restrict__ out)          // (B, NH)
{
    __shared__ float lds_ang[WAVES_PER_BLOCK][ROWS_PER_WAVE][16];

    const int tid    = threadIdx.x;
    const int wave   = tid >> 5;
    const int lane   = tid & 31;
    const int laneM  = lane & 15;
    const int laneHi = lane >> 4;
    const long rowBase = (long)blockIdx.x * ROWS_PER_BLOCK + (long)wave * ROWS_PER_WAVE;

    // ---------------- Stage 1: angles GEMM via WMMA ----------------
    // A tile t: rows rowBase + t*16 + laneM ; B = Wang^T, column N = laneM (zero-masked for N>=6)
    const float* aRow0 = sv + (rowBase + 0  + laneM) * (long)IN_DIM;
    const float* aRow1 = sv + (rowBase + 16 + laneM) * (long)IN_DIM;
    const int   bValid = (laneM < NQ);
    const float bMask  = bValid ? 1.0f : 0.0f;
    const float* bRow  = Wang + (long)(bValid ? laneM : 0) * IN_DIM;

    v8f acc0 = {};
    v8f acc1 = {};

    #pragma unroll 2
    for (int kc = 0; kc < IN_DIM; kc += 32) {
        const int k0 = kc + laneHi * 8;   // 32B aligned

        const float4 a0lo0 = *(const float4*)(aRow0 + k0);
        const float4 a0lo1 = *(const float4*)(aRow0 + k0 + 4);
        const float4 a0hi0 = *(const float4*)(aRow0 + k0 + 16);
        const float4 a0hi1 = *(const float4*)(aRow0 + k0 + 20);

        const float4 a1lo0 = *(const float4*)(aRow1 + k0);
        const float4 a1lo1 = *(const float4*)(aRow1 + k0 + 4);
        const float4 a1hi0 = *(const float4*)(aRow1 + k0 + 16);
        const float4 a1hi1 = *(const float4*)(aRow1 + k0 + 20);

        const float4 blo0  = *(const float4*)(bRow + k0);
        const float4 blo1  = *(const float4*)(bRow + k0 + 4);
        const float4 bhi0  = *(const float4*)(bRow + k0 + 16);
        const float4 bhi1  = *(const float4*)(bRow + k0 + 20);

        v16h A0, A1, Bm;
        const float* f;
        f = (const float*)&a0lo0; A0[0]=(_Float16)f[0]; A0[1]=(_Float16)f[1]; A0[2]=(_Float16)f[2]; A0[3]=(_Float16)f[3];
        f = (const float*)&a0lo1; A0[4]=(_Float16)f[0]; A0[5]=(_Float16)f[1]; A0[6]=(_Float16)f[2]; A0[7]=(_Float16)f[3];
        f = (const float*)&a0hi0; A0[8]=(_Float16)f[0]; A0[9]=(_Float16)f[1]; A0[10]=(_Float16)f[2]; A0[11]=(_Float16)f[3];
        f = (const float*)&a0hi1; A0[12]=(_Float16)f[0]; A0[13]=(_Float16)f[1]; A0[14]=(_Float16)f[2]; A0[15]=(_Float16)f[3];

        f = (const float*)&a1lo0; A1[0]=(_Float16)f[0]; A1[1]=(_Float16)f[1]; A1[2]=(_Float16)f[2]; A1[3]=(_Float16)f[3];
        f = (const float*)&a1lo1; A1[4]=(_Float16)f[0]; A1[5]=(_Float16)f[1]; A1[6]=(_Float16)f[2]; A1[7]=(_Float16)f[3];
        f = (const float*)&a1hi0; A1[8]=(_Float16)f[0]; A1[9]=(_Float16)f[1]; A1[10]=(_Float16)f[2]; A1[11]=(_Float16)f[3];
        f = (const float*)&a1hi1; A1[12]=(_Float16)f[0]; A1[13]=(_Float16)f[1]; A1[14]=(_Float16)f[2]; A1[15]=(_Float16)f[3];

        f = (const float*)&blo0;  Bm[0]=(_Float16)(f[0]*bMask); Bm[1]=(_Float16)(f[1]*bMask); Bm[2]=(_Float16)(f[2]*bMask); Bm[3]=(_Float16)(f[3]*bMask);
        f = (const float*)&blo1;  Bm[4]=(_Float16)(f[0]*bMask); Bm[5]=(_Float16)(f[1]*bMask); Bm[6]=(_Float16)(f[2]*bMask); Bm[7]=(_Float16)(f[3]*bMask);
        f = (const float*)&bhi0;  Bm[8]=(_Float16)(f[0]*bMask); Bm[9]=(_Float16)(f[1]*bMask); Bm[10]=(_Float16)(f[2]*bMask); Bm[11]=(_Float16)(f[3]*bMask);
        f = (const float*)&bhi1;  Bm[12]=(_Float16)(f[0]*bMask); Bm[13]=(_Float16)(f[1]*bMask); Bm[14]=(_Float16)(f[2]*bMask); Bm[15]=(_Float16)(f[3]*bMask);

        acc0 = __builtin_amdgcn_wmma_f32_16x16x32_f16(false, A0, false, Bm, (short)0, acc0, false, false);
        acc1 = __builtin_amdgcn_wmma_f32_16x16x32_f16(false, A1, false, Bm, (short)0, acc1, false, false);
    }

    // bias (per column N) and transpose through LDS into per-lane rows
    const float biasN = bValid ? bang[laneM] : 0.0f;
    #pragma unroll
    for (int g = 0; g < 8; ++g) {
        lds_ang[wave][ 0 + g + 8 * laneHi][laneM] = acc0[g] + biasN;
        lds_ang[wave][16 + g + 8 * laneHi][laneM] = acc1[g] + biasN;
    }
    __syncthreads();

    float ang[NQ];
    #pragma unroll
    for (int q = 0; q < NQ; ++q) ang[q] = lds_ang[wave][lane][q];

    // ---------------- Stage 2: 6-qubit circuit, one row per lane ----------------
    // RY layer on |0..0>: real product state. wire w <-> bit (5-w) of index i.
    float cw[NQ], sw[NQ];
    #pragma unroll
    for (int w = 0; w < NQ; ++w) {
        const float h = 0.5f * ang[w];
        cw[w] = __cosf(h);
        sw[w] = __sinf(h);
    }

    float re[QDIM], im[QDIM];
    #pragma unroll
    for (int i = 0; i < QDIM; ++i) {
        float p = 1.0f;
        #pragma unroll
        for (int w = 0; w < NQ; ++w)
            p *= ((i >> (5 - w)) & 1) ? sw[w] : cw[w];
        re[i] = p;
        im[i] = 0.0f;
    }

    #pragma unroll
    for (int l = 0; l < NL; ++l) {
        #pragma unroll
        for (int w = 0; w < NQ; ++w) {
            const int gidx = (l * NQ + w) * 3;
            const float phi   = qw[gidx + 0];
            const float theta = qw[gidx + 1];
            const float omega = qw[gidx + 2];
            const float ct = __cosf(0.5f * theta), st = __sinf(0.5f * theta);
            const float hp = 0.5f * (phi + omega), hm = 0.5f * (phi - omega);
            const float cp = __cosf(hp), sp = __sinf(hp);
            const float cm = __cosf(hm), sm = __sinf(hm);
            // U00 = conj(ep)*ct, U01 = -em*st, U10 = conj(em)*st, U11 = ep*ct
            const float u00r =  cp * ct, u00i = -sp * ct;
            const float u01r = -cm * st, u01i = -sm * st;
            const float u10r =  cm * st, u10i = -sm * st;
            const float u11r =  cp * ct, u11i =  sp * ct;

            const int bit = 1 << (5 - w);
            #pragma unroll
            for (int i0 = 0; i0 < QDIM; ++i0) {
                if (i0 & bit) continue;
                const int i1 = i0 | bit;
                const float a0r = re[i0], a0i = im[i0];
                const float a1r = re[i1], a1i = im[i1];
                re[i0] = u00r * a0r - u00i * a0i + u01r * a1r - u01i * a1i;
                im[i0] = u00r * a0i + u00i * a0r + u01r * a1i + u01i * a1r;
                re[i1] = u10r * a0r - u10i * a0i + u11r * a1r - u11i * a1i;
                im[i1] = u10r * a0i + u10i * a0r + u11r * a1i + u11i * a1r;
            }
        }
        // CNOT ring: ctrl=w, tgt=(w+r)%NQ — pure register permutation (free)
        const int r = (l % (NQ - 1)) + 1;
        #pragma unroll
        for (int w = 0; w < NQ; ++w) {
            const int cbit = 1 << (5 - w);
            const int tbit = 1 << (5 - ((w + r) % NQ));
            #pragma unroll
            for (int i = 0; i < QDIM; ++i) {
                if ((i & cbit) && !(i & tbit)) {
                    const int j = i | tbit;
                    float t;
                    t = re[i]; re[i] = re[j]; re[j] = t;
                    t = im[i]; im[i] = im[j]; im[j] = t;
                }
            }
        }
    }

    // ---------------- Stage 3: probs -> qfeat -> head -> 1 + 0.5*tanh ----------------
    float qf[NQ];
    #pragma unroll
    for (int q = 0; q < NQ; ++q) qf[q] = 0.0f;
    #pragma unroll
    for (int i = 0; i < QDIM; ++i) {
        const float p = re[i] * re[i] + im[i] * im[i];
        #pragma unroll
        for (int q = 0; q < NQ; ++q)
            qf[q] += ((i >> (5 - q)) & 1) ? -p : p;
    }

    float ovals[NH];
    #pragma unroll
    for (int h = 0; h < NH; ++h) {
        float z = bpost[h];
        #pragma unroll
        for (int q = 0; q < NQ; ++q)
            z = fmaf(qf[q], Wpost[h * NQ + q], z);
        ovals[h] = 1.0f + 0.5f * fast_tanhf(z);
    }

    float* orow = out + (rowBase + lane) * (long)NH;
    #pragma unroll
    for (int h = 0; h < NH; h += 4) {
        float4 v = make_float4(ovals[h], ovals[h + 1], ovals[h + 2], ovals[h + 3]);
        *(float4*)(orow + h) = v;
    }
}

extern "C" void kernel_launch(void* const* d_in, const int* in_sizes, int n_in,
                              void* d_out, int out_size, void* d_ws, size_t ws_size,
                              hipStream_t stream) {
    const float* sv    = (const float*)d_in[0];  // summary_vec (B, IN_DIM)
    const float* Wang  = (const float*)d_in[1];  // W_angles   (NQ, IN_DIM)
    const float* bang  = (const float*)d_in[2];  // b_angles   (NQ,)
    const float* Wpost = (const float*)d_in[3];  // W_post     (NH, NQ)
    const float* bpost = (const float*)d_in[4];  // b_post     (NH,)
    const float* qw    = (const float*)d_in[5];  // q_weights  (NL, NQ, 3)
    float* out = (float*)d_out;

    const int grid = BATCH / ROWS_PER_BLOCK;     // 32768 / 64 = 512 blocks
    qagate_fused_kernel<<<grid, THREADS_PER_BLOCK, 0, stream>>>(
        sv, Wang, bang, Wpost, bpost, qw, out);
}
